// MoELayer_81415400063491
// MI455X (gfx1250) — compile-verified, matrix operand order checked by reading, not run on hardware
//
#include <hip/hip_runtime.h>
#include <hip/hip_bf16.h>
#include <math.h>

// ---------------------------------------------------------------------------
// MoE layer for MI455X (gfx1250, wave32, WMMA).
// Routed-expert compute with bf16 WMMA (16x16x32, f32 accum). The routed
// GEMMs are ~275 GFLOP vs ~300 MB of HBM traffic (~900 FLOP/byte) -> compute
// bound, so we maximize matrix-pipe work per byte: 32-token M tiles give 2x
// weight-fragment reuse (each B fragment feeds 2 WMMAs), per-expert weights
// (32 MB) live in the 192 MB L2 across the ~64 blocks of that expert.
// ---------------------------------------------------------------------------

typedef __attribute__((ext_vector_type(16))) __bf16 v16bf;
typedef __attribute__((ext_vector_type(8)))  __bf16 v8bf;
typedef __attribute__((ext_vector_type(8)))  float  v8f;

#define T_TOK   8192      // B*S
#define DIM     1024      // D
#define NEXP    8         // E
#define HID     4096      // H
#define TILE_M  32        // tokens per block tile (2 WMMA M-subtiles)
#define CHUNK_H 256       // H columns per fused chunk
#define THREADS 256       // 8 wave32s

// ISA 7.12.2: 16-bit A/B fragment K index for element j of the per-lane v16bf.
// lanes 0-15 (half=0): K = {0..7, 16..23};  lanes 16-31 (half=1): {8..15, 24..31}
__device__ __forceinline__ int kmap(int j, int half) {
    return (j & 7) + 8 * half + 16 * (j >> 3);
}

// Build a 16-bit A/B fragment from a row-major bf16 row in LDS: two
// contiguous 16-byte runs -> 2x ds_load_b128 per fragment.
__device__ __forceinline__ v16bf frag_from_row(const __bf16* row, int kc, int half) {
    v8bf lo = *(const v8bf*)(row + kc + 8 * half);
    v8bf hi = *(const v8bf*)(row + kc + 16 + 8 * half);
    v16bf f;
#pragma unroll
    for (int j = 0; j < 8; ++j) { f[j] = lo[j]; f[j + 8] = hi[j]; }
    return f;
}

__device__ __forceinline__ float gelu_exact(float v) {
    return 0.5f * v * (1.0f + erff(v * 0.70710678118654752f));
}

// ---------------------------------------------------------------------------
__global__ void moe_init(int* __restrict__ counts, float* __restrict__ psum) {
    int i = threadIdx.x;
    if (i < NEXP) { counts[i] = 0; psum[i] = 0.0f; }
}

__global__ void moe_zero_out(float4* __restrict__ out) {
    size_t i = (size_t)blockIdx.x * blockDim.x + threadIdx.x;
    out[i] = make_float4(0.f, 0.f, 0.f, 0.f);
}

// ---------------------------------------------------------------------------
// Router: one wave32 per token. 8 logits per token; lane-strided dot over D,
// shfl-xor reduction, softmax, top-2 (first-index tie-break like lax.top_k),
// normalized gates; builds per-expert compact token lists via atomics.
__global__ __launch_bounds__(THREADS)
void moe_router(const float* __restrict__ x,
                const float* __restrict__ rw,   // [D, E]
                const float* __restrict__ rb,   // [E]
                int*   __restrict__ counts,     // [E]
                int*   __restrict__ etok,       // [E, T]
                float* __restrict__ egate,      // [E, T]
                float* __restrict__ psum)       // [E] sum of probs over tokens
{
    const int wave = threadIdx.x >> 5;
    const int lane = threadIdx.x & 31;
    const int t    = blockIdx.x * (THREADS / 32) + wave;

    float acc[NEXP];
#pragma unroll
    for (int e = 0; e < NEXP; ++e) acc[e] = 0.0f;

    const float* xt = x + (size_t)t * DIM;
    for (int d = lane; d < DIM; d += 32) {
        float xv = xt[d];
        const float4* rwp = (const float4*)(rw + (size_t)d * NEXP);
        float4 a = rwp[0], b = rwp[1];
        acc[0] += xv * a.x; acc[1] += xv * a.y;
        acc[2] += xv * a.z; acc[3] += xv * a.w;
        acc[4] += xv * b.x; acc[5] += xv * b.y;
        acc[6] += xv * b.z; acc[7] += xv * b.w;
    }
#pragma unroll
    for (int e = 0; e < NEXP; ++e)
#pragma unroll
        for (int off = 16; off > 0; off >>= 1)
            acc[e] += __shfl_xor(acc[e], off, 32);

    if (lane == 0) {
        float p[NEXP];
        float mx = -3.4e38f;
#pragma unroll
        for (int e = 0; e < NEXP; ++e) {
            p[e] = acc[e] + rb[e];
            mx = fmaxf(mx, p[e]);
        }
        float s = 0.0f;
#pragma unroll
        for (int e = 0; e < NEXP; ++e) { p[e] = __expf(p[e] - mx); s += p[e]; }
        float inv = 1.0f / s;
#pragma unroll
        for (int e = 0; e < NEXP; ++e) {
            p[e] *= inv;
            atomicAdd(&psum[e], p[e]);
        }
        // top-2, first-occurrence tie break
        int e0 = 0;
#pragma unroll
        for (int e = 1; e < NEXP; ++e) if (p[e] > p[e0]) e0 = e;
        int e1 = (e0 == 0) ? 1 : 0;
#pragma unroll
        for (int e = 0; e < NEXP; ++e)
            if (e != e0 && p[e] > p[e1]) e1 = e;
        float w0 = p[e0], w1 = p[e1];
        float inv2 = 1.0f / (w0 + w1);
        w0 *= inv2; w1 *= inv2;
        int pos0 = atomicAdd(&counts[e0], 1);
        etok[e0 * T_TOK + pos0] = t; egate[e0 * T_TOK + pos0] = w0;
        int pos1 = atomicAdd(&counts[e1], 1);
        etok[e1 * T_TOK + pos1] = t; egate[e1 * T_TOK + pos1] = w1;
    }
}

__global__ void moe_lbl(const float* __restrict__ psum, float* __restrict__ dst) {
    if (threadIdx.x == 0) {
        float loss = 0.0f;
#pragma unroll
        for (int e = 0; e < NEXP; ++e) {
            float m = psum[e] * (1.0f / (float)T_TOK);
            loss += m * logf(m);
        }
        dst[0] = (float)NEXP * loss;
    }
}

// ---------------------------------------------------------------------------
// Fused expert FFN: per block, a 32-token tile (2 WMMA M-subtiles) of one
// expert. For each 256-wide H chunk:
//   inner:  h[32,256] = X @ w1 chunk  (wave owns 32 h-cols; B reused 2x)
//   gelu -> bf16 chunk in LDS
//   outer:  y[32,128/wave] += h_chunk @ w2 chunk  (B reused 2x)
// epilogue: out[token] += gate * (y + b2) via f32 global atomics.
__global__ __launch_bounds__(THREADS)
void moe_expert_ffn(const float* __restrict__ x,
                    const float* __restrict__ w1,   // [E, D, H]
                    const float* __restrict__ b1,   // [E, H]
                    const float* __restrict__ w2,   // [E, H, D]
                    const float* __restrict__ b2,   // [E, D]
                    const int*   __restrict__ counts,
                    const int*   __restrict__ etok,
                    const float* __restrict__ egate,
                    float* __restrict__ out)        // [T, D]
{
    __shared__ __bf16 xs[TILE_M][DIM];       // 64 KB: token tile, bf16
    __shared__ __bf16 hs[TILE_M][CHUNK_H];   // 16 KB: gelu(h) chunk, bf16
    __shared__ int    toks[TILE_M];
    __shared__ float  gates[TILE_M];

    const int e    = blockIdx.x / (T_TOK / TILE_M);
    const int tile = blockIdx.x % (T_TOK / TILE_M);
    const int cnt  = counts[e];
    if (tile * TILE_M >= cnt) return;        // uniform exit (over-provisioned grid)

    const int tid  = threadIdx.x;
    const int wave = tid >> 5;
    const int lane = tid & 31;
    const int half = lane >> 4;
    const int idx  = lane & 15;

    if (tid < TILE_M) {
        int r = tile * TILE_M + tid;
        if (r < cnt) { toks[tid] = etok[e * T_TOK + r]; gates[tid] = egate[e * T_TOK + r]; }
        else         { toks[tid] = -1;                  gates[tid] = 0.0f; }
    }
    __syncthreads();

    // Stage X tile to LDS as bf16 (float4 global loads, packed bf16 stores)
    for (int i = tid; i < TILE_M * (DIM / 4); i += THREADS) {
        int m  = i / (DIM / 4);
        int d4 = i % (DIM / 4);
        int t  = toks[m];
        float4 v = (t >= 0) ? ((const float4*)(x + (size_t)t * DIM))[d4]
                            : make_float4(0.f, 0.f, 0.f, 0.f);
        __bf16* dst = &xs[m][d4 * 4];
        dst[0] = (__bf16)v.x; dst[1] = (__bf16)v.y;
        dst[2] = (__bf16)v.z; dst[3] = (__bf16)v.w;
    }
    __syncthreads();

    const float* w1e = w1 + (size_t)e * DIM * HID;
    const float* w2e = w2 + (size_t)e * HID * DIM;
    const float* b1e = b1 + (size_t)e * HID;
    const float* b2e = b2 + (size_t)e * DIM;

    v8f acc[16];                     // y accumulator: [mt(2)][nt(8)] x 16 cols
#pragma unroll
    for (int i = 0; i < 16; ++i)
#pragma unroll
        for (int r = 0; r < 8; ++r) acc[i][r] = 0.0f;

    for (int hc = 0; hc < HID; hc += CHUNK_H) {
        // ---- inner GEMM: h[32 rows, 32 cols owned by this wave] over K = D
        v8f hacc[4];                 // [mt(2)][nt(2)]
#pragma unroll
        for (int i = 0; i < 4; ++i)
#pragma unroll
            for (int r = 0; r < 8; ++r) hacc[i][r] = 0.0f;

        const int col0 = hc + wave * 32;
        if (hc + CHUNK_H < HID)   // nudge next weight chunk toward L2 (global_prefetch_b8)
            __builtin_prefetch(&w1e[(size_t)(hc + CHUNK_H) + wave * 32 + idx], 0, 1);

        for (int kc = 0; kc < DIM; kc += 32) {
            v16bf am0 = frag_from_row(&xs[idx][0],      kc, half);
            v16bf am1 = frag_from_row(&xs[16 + idx][0], kc, half);
#pragma unroll
            for (int nt = 0; nt < 2; ++nt) {
                const int col = col0 + nt * 16 + idx;
                v16bf bf;
#pragma unroll
                for (int j = 0; j < 16; ++j)
                    bf[j] = (__bf16)w1e[(size_t)(kc + kmap(j, half)) * HID + col];
                hacc[0 * 2 + nt] = __builtin_amdgcn_wmma_f32_16x16x32_bf16(
                    false, am0, false, bf, (short)0, hacc[0 * 2 + nt], false, false);
                hacc[1 * 2 + nt] = __builtin_amdgcn_wmma_f32_16x16x32_bf16(
                    false, am1, false, bf, (short)0, hacc[1 * 2 + nt], false, false);
            }
        }
        // bias + exact GeLU -> bf16 chunk in LDS
#pragma unroll
        for (int nt = 0; nt < 2; ++nt) {
            const int col  = col0 + nt * 16 + idx;
            const int lcol = wave * 32 + nt * 16 + idx;
            const float bias = b1e[col];
#pragma unroll
            for (int mt = 0; mt < 2; ++mt)
#pragma unroll
                for (int r = 0; r < 8; ++r) {
                    int m = mt * 16 + r + 8 * half;
                    hs[m][lcol] = (__bf16)gelu_exact(hacc[mt * 2 + nt][r] + bias);
                }
        }
        __syncthreads();

        // ---- outer GEMM: y[32 rows, 128 cols owned by this wave] += h @ w2
        for (int ks = 0; ks < CHUNK_H; ks += 32) {
            v16bf am0 = frag_from_row(&hs[idx][0],      ks, half);
            v16bf am1 = frag_from_row(&hs[16 + idx][0], ks, half);
#pragma unroll
            for (int nt = 0; nt < 8; ++nt) {
                const int dcol = wave * 128 + nt * 16 + idx;
                v16bf bf;
#pragma unroll
                for (int j = 0; j < 16; ++j)
                    bf[j] = (__bf16)w2e[(size_t)(hc + ks + kmap(j, half)) * DIM + dcol];
                acc[0 * 8 + nt] = __builtin_amdgcn_wmma_f32_16x16x32_bf16(
                    false, am0, false, bf, (short)0, acc[0 * 8 + nt], false, false);
                acc[1 * 8 + nt] = __builtin_amdgcn_wmma_f32_16x16x32_bf16(
                    false, am1, false, bf, (short)0, acc[1 * 8 + nt], false, false);
            }
        }
        __syncthreads();   // hs rewritten next chunk
    }

    // ---- epilogue: out[t] += gate * (y + b2); atomics (token in 2 expert lists)
#pragma unroll
    for (int nt = 0; nt < 8; ++nt) {
        const int dcol = wave * 128 + nt * 16 + idx;
        const float bias = b2e[dcol];
#pragma unroll
        for (int mt = 0; mt < 2; ++mt)
#pragma unroll
            for (int r = 0; r < 8; ++r) {
                int m = mt * 16 + r + 8 * half;
                int t = toks[m];
                if (t >= 0) {
                    float v = gates[m] * (acc[mt * 8 + nt][r] + bias);
                    atomicAdd(&out[(size_t)t * DIM + dcol], v);
                }
            }
    }
}

// ---------------------------------------------------------------------------
extern "C" void kernel_launch(void* const* d_in, const int* in_sizes, int n_in,
                              void* d_out, int out_size, void* d_ws, size_t ws_size,
                              hipStream_t stream) {
    const float* x  = (const float*)d_in[0];   // [B,S,D]
    const float* rw = (const float*)d_in[1];   // [D,E]
    const float* rb = (const float*)d_in[2];   // [E]
    const float* w1 = (const float*)d_in[3];   // [E,D,H]
    const float* b1 = (const float*)d_in[4];   // [E,H]
    const float* w2 = (const float*)d_in[5];   // [E,H,D]
    const float* b2 = (const float*)d_in[6];   // [E,D]
    float* out = (float*)d_out;                // [T*D] floats + 1 loss scalar

    // workspace layout (all 4-byte units): counters | psum | token lists | gates
    int*   counts = (int*)d_ws;                                // [E]
    float* psum   = (float*)d_ws + NEXP;                       // [E]
    int*   etok   = (int*)d_ws + 2 * NEXP;                     // [E*T]
    float* egate  = (float*)d_ws + 2 * NEXP + NEXP * T_TOK;    // [E*T]

    moe_init<<<1, 64, 0, stream>>>(counts, psum);
    moe_zero_out<<<(T_TOK * DIM / 4) / THREADS, THREADS, 0, stream>>>((float4*)out);
    moe_router<<<T_TOK / (THREADS / 32), THREADS, 0, stream>>>(
        x, rw, rb, counts, etok, egate, psum);
    moe_lbl<<<1, 32, 0, stream>>>(psum, out + (size_t)T_TOK * DIM);
    moe_expert_ffn<<<NEXP * (T_TOK / TILE_M), THREADS, 0, stream>>>(
        x, w1, b1, w2, b2, counts, etok, egate, out);
}
